// PiecewiseDiscontinuous_22780506538400
// MI455X (gfx1250) — matrix operand order repaired and checked by reading, not compiled, not run on hardware
//
#include <hip/hip_runtime.h>

typedef __attribute__((ext_vector_type(16))) _Float16 v16h;
typedef __attribute__((ext_vector_type(8)))  _Float16 v8h;
typedef __attribute__((ext_vector_type(2)))  _Float16 v2h;
typedef __attribute__((ext_vector_type(8)))  float    v8f;

#define BATCH 512
#define FIN   256
#define FOUT  256
#define NP    4
#define NSEG  32
#define KSEG  (NSEG * NP)      // 128 K-values per input feature
#define KTOT  (FIN * KSEG)     // 32768 total contraction length
#define NKC   (KTOT / 32)      // 1024 K-chunks of 32 (one wmma each)

// ---------------------------------------------------------------------------
// Kernel 1: expand basis into sparse A matrix (f16), row = b, col = f*128+seg*4+n
// Row segment of 128 halves (256B) is built in registers: 15 zero uint4s + one
// uint4 carrying the 4 basis halves at (seg*4) -> single typed store stream.
// ---------------------------------------------------------------------------
__global__ void build_A_kernel(const float* __restrict__ x, _Float16* __restrict__ A)
{
    int idx = blockIdx.x * blockDim.x + threadIdx.x;   // = b*FIN + f
    float t = x[idx];

    // segment index: trunc-toward-zero then clamp (matches astype(int32)+clip)
    int seg = (int)((t + 1.0f) * 16.0f);
    seg = seg < 0 ? 0 : (seg > NSEG - 1 ? NSEG - 1 : seg);

    // local coordinate in [-1, 1]
    float xmin = (float)seg * (2.0f / NSEG) - 1.0f;
    float u = 32.0f * (t - xmin) - 1.0f;

    // Lagrange basis at Chebyshev-Lobatto nodes {-1, -0.5, 0.5, 1}
    float um1 = u - 1.0f, up1 = u + 1.0f, umh = u - 0.5f, uph = u + 0.5f;
    float b0 = uph * umh * um1 * (-1.0f / 1.50f);
    float b1 = up1 * umh * um1 * ( 1.0f / 0.75f);
    float b2 = up1 * uph * um1 * (-1.0f / 0.75f);
    float b3 = up1 * uph * umh * ( 1.0f / 1.50f);

    v2h h01 = { (_Float16)b0, (_Float16)b1 };
    v2h h23 = { (_Float16)b2, (_Float16)b3 };
    unsigned u01 = __builtin_bit_cast(unsigned, h01);
    unsigned u23 = __builtin_bit_cast(unsigned, h23);

    uint4 z   = make_uint4(0u, 0u, 0u, 0u);
    uint4 nzq = (seg & 1) ? make_uint4(0u, 0u, u01, u23)
                          : make_uint4(u01, u23, 0u, 0u);
    int qi = seg >> 1;                       // which 16B quad holds the 8B payload

    uint4* rowq = (uint4*)(A + (size_t)idx * KSEG);
#pragma unroll
    for (int i = 0; i < 16; ++i)
        rowq[i] = (i == qi) ? nzq : z;
}

// ---------------------------------------------------------------------------
// Kernel 2: convert w (FOUT,FIN,128 f32) into B-fragment-tiled f16:
//   Wt[kc][ot][lane][e] , lane = 16*half + (o%16), element e -> K = kc*32+16*half+e
// so the GEMM reads each lane's B fragment as 2 contiguous b128 loads.
// ---------------------------------------------------------------------------
__device__ inline v8h cvt8(float4 a, float4 b)
{
    v8h r = { (_Float16)a.x, (_Float16)a.y, (_Float16)a.z, (_Float16)a.w,
              (_Float16)b.x, (_Float16)b.y, (_Float16)b.z, (_Float16)b.w };
    return r;
}

__global__ void build_W_kernel(const float* __restrict__ w, _Float16* __restrict__ Wt)
{
    int idx  = blockIdx.x * blockDim.x + threadIdx.x;  // (kc, ot, lane)
    int lane = idx & 31;
    int ot   = (idx >> 5) & 15;
    int kc   = idx >> 9;
    int o    = ot * 16 + (lane & 15);
    int h    = lane >> 4;
    int f    = kc >> 2;                 // 4 K-chunks per input feature
    int kk   = (kc & 3) * 32 + 16 * h;  // this lane's 16 contiguous K values

    const float4* wq = (const float4*)(w + ((size_t)o * FIN + f) * KSEG + kk);
    v8h lo = cvt8(wq[0], wq[1]);
    v8h hi = cvt8(wq[2], wq[3]);

    v8h* dst = (v8h*)(Wt + (size_t)idx * 16);
    dst[0] = lo;
    dst[1] = hi;
}

// ---------------------------------------------------------------------------
// Kernel 3: C(512x256) = A(512x32768) x Wt(32768x256), f16 in / f32 accumulate.
// One wave per 16x16 output tile; sparse-chunk skip via wave ballot.
// ---------------------------------------------------------------------------
__global__ void __launch_bounds__(128)
gemm_kernel(const _Float16* __restrict__ A, const _Float16* __restrict__ Wt,
            float* __restrict__ out)
{
    int wave = threadIdx.x >> 5;
    int lane = threadIdx.x & 31;
    int tile = blockIdx.x * 4 + wave;   // 0..511
    int bt   = tile & 31;               // batch tile (16 rows)
    int ot   = tile >> 5;               // output-feature tile (16 cols)
    int m    = lane & 15;
    int h    = lane >> 4;

    // A fragment (contiguous order, matches B): e=0..15 -> K = kc*32 + 16h + e
    const uint4* arow = (const uint4*)(A + (size_t)(bt * 16 + m) * KTOT) + 2 * h;
    // B fragment: pre-tiled, 2 consecutive uint4 per lane per chunk
    const uint4* brow = (const uint4*)Wt + ((size_t)ot * 32 + lane) * 2;

    v8f acc = {};
    for (int kc = 0; kc < NKC; ++kc) {
        uint4 a0 = arow[(size_t)kc * 4];
        uint4 a1 = arow[(size_t)kc * 4 + 1];
        __builtin_prefetch((const void*)(arow + (size_t)(kc + 8) * 4), 0, 0);

        bool nz = (a0.x | a0.y | a0.z | a0.w | a1.x | a1.y | a1.z | a1.w) != 0u;
        if (__any(nz)) {                        // wave-uniform: EXEC stays full
            uint4 b0 = brow[(size_t)kc * 1024];
            uint4 b1 = brow[(size_t)kc * 1024 + 1];

            v8h alo = __builtin_bit_cast(v8h, a0);
            v8h ahi = __builtin_bit_cast(v8h, a1);
            v8h blo = __builtin_bit_cast(v8h, b0);
            v8h bhi = __builtin_bit_cast(v8h, b1);
            v16h av = __builtin_shufflevector(alo, ahi, 0,1,2,3,4,5,6,7,8,9,10,11,12,13,14,15);
            v16h bv = __builtin_shufflevector(blo, bhi, 0,1,2,3,4,5,6,7,8,9,10,11,12,13,14,15);

            acc = __builtin_amdgcn_wmma_f32_16x16x32_f16(
                      false, av, false, bv, (short)0, acc, false, false);
        }
    }

    // C/D layout: lane n = lane%16, VGPR r holds row m = r + 8*(lane/16)
    int o   = ot * 16 + m;
    int b0r = bt * 16 + 8 * h;
#pragma unroll
    for (int r = 0; r < 8; ++r)
        out[(size_t)(b0r + r) * FOUT + o] = acc[r];
}

// ---------------------------------------------------------------------------
extern "C" void kernel_launch(void* const* d_in, const int* in_sizes, int n_in,
                              void* d_out, int out_size, void* d_ws, size_t ws_size,
                              hipStream_t stream)
{
    (void)in_sizes; (void)n_in; (void)out_size; (void)ws_size;
    const float* x = (const float*)d_in[0];   // (512, 256)
    const float* w = (const float*)d_in[1];   // (256, 256, 128)
    float* out = (float*)d_out;               // (512, 256)

    _Float16* A  = (_Float16*)d_ws;                                    // 32 MB
    _Float16* Wt = (_Float16*)((char*)d_ws + (size_t)BATCH * KTOT * 2); // 16.7 MB

    build_A_kernel<<<(BATCH * FIN) / 256, 256, 0, stream>>>(x, A);
    build_W_kernel<<<(NKC * 16 * 32) / 256, 256, 0, stream>>>(w, Wt);
    gemm_kernel<<<(BATCH / 16) * (FOUT / 16) / 4, 128, 0, stream>>>(A, Wt, out);
}
